// AttentionBlock_90924457656471
// MI455X (gfx1250) — matrix operand order repaired and checked by reading, not compiled, not run on hardware
//
#include <hip/hip_runtime.h>
#include <hip/hip_bf16.h>

typedef __attribute__((ext_vector_type(16))) _Float16 v16h;
typedef __attribute__((ext_vector_type(8)))  _Float16 v8h;
typedef __attribute__((ext_vector_type(8)))  float    v8f;
typedef __attribute__((ext_vector_type(4))) unsigned int u32x4;
typedef __attribute__((ext_vector_type(8))) int          i32x8;
typedef __attribute__((ext_vector_type(4))) int          i32x4;

#define WMMA_F16(A, B, C) \
  __builtin_amdgcn_wmma_f32_16x16x32_f16(false, (A), false, (B), (short)0, (C), false, false)

#if defined(__AMDGCN__) && __has_builtin(__builtin_amdgcn_tensor_load_to_lds)
#define USE_TDM 1
#if __has_include(<hip/amd_detail/amd_gfx1250_TDM.h>)
#define TDM_SIX_ARGS 1
#else
#define TDM_SIX_ARGS 0
#endif
#else
#define USE_TDM 0
#endif

// ---------------- problem constants ----------------
constexpr int Bz = 16, Cc = 256, S = 1024;          // B, C, H*W
constexpr int NH = 8, DK = 32, DA = 256;            // heads, head dim, attn dim
constexpr int TD = 512, CD = 128, IN_DIM = 640;     // t_emb, cond, concat

// ---------------- WMMA fragment loaders (wave32, contiguous k = 16*hf + i order) -------
// A (16xK=32, f16): lane m=lane&15 holds row M=m; halfs i -> K = 16*hf + i
// (contiguous run, hf = lane>>4)
__device__ __forceinline__ v16h load_a_frag(const _Float16* base, int ld, int lane) {
  const int m = lane & 15, hf = lane >> 4;
  const _Float16* row = base + (size_t)m * ld;
  v8h lo = *(const v8h*)(row + 16 * hf);
  v8h hi = *(const v8h*)(row + 16 * hf + 8);
  v16h a;
#pragma unroll
  for (int i = 0; i < 8; ++i) { a[i] = lo[i]; a[8 + i] = hi[i]; }
  return a;
}

// B (K=32 x 16, f16), source stored as src[n*ld + k] (row n = output column):
// lane holds column N=lane&15, halfs i -> K = 16*hf + i  (contiguous run)
__device__ __forceinline__ v16h load_b_cont(const _Float16* base, int ld, int lane) {
  const int n = lane & 15, hf = lane >> 4;
  const _Float16* p = base + (size_t)n * ld + 16 * hf;
  v8h lo = *(const v8h*)(p);
  v8h hi = *(const v8h*)(p + 8);
  v16h b;
#pragma unroll
  for (int i = 0; i < 8; ++i) { b[i] = lo[i]; b[8 + i] = hi[i]; }
  return b;
}

// B (K=32 x 16, f16), source stored row-major src[k*ld + n] (e.g. V tile)
__device__ __forceinline__ v16h load_b_rowmajor(const _Float16* base, int ld, int lane) {
  const int n = lane & 15, hf = lane >> 4;
  v16h b;
#pragma unroll
  for (int i = 0; i < 16; ++i) b[i] = base[(size_t)(16 * hf + i) * ld + n];
  return b;
}

#if USE_TDM
// 1-D TDM copy: nelem f16 elements, global -> LDS (contiguous), per ISA D#.
__device__ __forceinline__ void tdm_load_1d_f16(unsigned lds_byte_addr,
                                                unsigned long long gaddr,
                                                unsigned nelem) {
  union { u32x4 v; unsigned w[4]; } g0 = {};
  g0.w[0] = 1u;                                        // count = 1 (valid D#)
  g0.w[1] = lds_byte_addr;                             // lds_addr [63:32]
  g0.w[2] = (unsigned)(gaddr & 0xFFFFFFFFull);         // global_addr lo
  g0.w[3] = (unsigned)((gaddr >> 32) & 0x1FFFFFFull)   // global_addr [56:32]
            | (2u << 30);                              // type = 2 ("image")
  union { i32x8 v; unsigned w[8]; } g1 = {};
  g1.w[0] = 1u << 16;                                  // data_size = 2 bytes
  g1.w[1] = (nelem & 0xFFFFu) << 16;                   // tensor_dim0 [79:48] lo
  g1.w[2] = ((nelem >> 16) & 0xFFFFu)                  // tensor_dim0 hi
            | (1u << 16);                              // tensor_dim1 = 1
  g1.w[3] = (nelem & 0xFFFFu) << 16;                   // tile_dim0 [127:112]
  i32x4 gz = {};
#if TDM_SIX_ARGS
  i32x8 gz8 = {};
  __builtin_amdgcn_tensor_load_to_lds(g0.v, g1.v, gz, gz, gz8, 0);
#else
  __builtin_amdgcn_tensor_load_to_lds(g0.v, g1.v, gz, gz, 0);
#endif
}
#endif

// ---------------- 1. adaLN projection: silu(concat) @ proj_w^T + proj_b ----
__global__ __launch_bounds__(512) void adaln_kernel(
    const float* __restrict__ t_emb, const float* __restrict__ cond,
    const float* __restrict__ pw, const float* __restrict__ pb,
    float* __restrict__ gamma, float* __restrict__ beta) {
  __shared__ float si[IN_DIM];
  const int b = blockIdx.x;
  for (int i = threadIdx.x; i < IN_DIM; i += 512) {
    float v = (i < TD) ? t_emb[b * TD + i] : cond[b * CD + (i - TD)];
    si[i] = v / (1.0f + __expf(-v));
  }
  __syncthreads();
  const int j = threadIdx.x;  // 0..511
  const float* wr = pw + (size_t)j * IN_DIM;
  float acc = pb[j];
  for (int i = 0; i < IN_DIM; ++i) acc = fmaf(si[i], wr[i], acc);
  if (j < Cc) gamma[b * Cc + j] = acc;
  else        beta[b * Cc + (j - Cc)] = acc;
}

// ---------------- 2. GroupNorm statistics ----------------------------------
__global__ __launch_bounds__(256) void gn_stats_kernel(
    const float* __restrict__ x, float* __restrict__ gnm, float* __restrict__ gnr) {
  __shared__ float ssum[256], ssq[256];
  const int blk = blockIdx.x;               // 0..127  (b*8+g)
  const int b = blk >> 3, g = blk & 7;
  const float* base = x + ((size_t)b * Cc + g * 32) * S;   // 32 contiguous chans
  float s = 0.f, q = 0.f;
  for (int i = threadIdx.x; i < 32 * S; i += 256) {
    float v = base[i]; s += v; q = fmaf(v, v, q);
  }
  ssum[threadIdx.x] = s; ssq[threadIdx.x] = q;
  __syncthreads();
  for (int off = 128; off > 0; off >>= 1) {
    if (threadIdx.x < off) {
      ssum[threadIdx.x] += ssum[threadIdx.x + off];
      ssq[threadIdx.x]  += ssq[threadIdx.x + off];
    }
    __syncthreads();
  }
  if (threadIdx.x == 0) {
    const float inv_n = 1.0f / (32.0f * S);
    float mean = ssum[0] * inv_n;
    float var  = ssq[0] * inv_n - mean * mean;
    gnm[blk] = mean;
    gnr[blk] = rsqrtf(var + 1e-6f);
  }
}

// ---------------- 3. normalize + adaLN modulate + transpose -> f16 (B,S,C) -
__global__ __launch_bounds__(256) void gn_norm_kernel(
    const float* __restrict__ x, const float* __restrict__ gnw,
    const float* __restrict__ gnb, const float* __restrict__ gamma,
    const float* __restrict__ beta, const float* __restrict__ gnm,
    const float* __restrict__ gnr, _Float16* __restrict__ xn) {
  const int b = blockIdx.x >> 8, c = blockIdx.x & 255;
  const int g = c >> 5;
  const float mean = gnm[b * 8 + g], rstd = gnr[b * 8 + g];
  const float sc = gnw[c] * rstd;
  const float sh = gnb[c] - mean * sc;
  const float ga = 1.0f + gamma[b * Cc + c];
  const float be = beta[b * Cc + c];
  const float A = sc * ga, Bc = fmaf(sh, ga, be);   // out = x*A + Bc
  const float* xr = x + ((size_t)b * Cc + c) * S;
  _Float16* xo = xn + (size_t)b * S * Cc + c;
  for (int s = threadIdx.x; s < S; s += 256)
    xo[(size_t)s * Cc] = (_Float16)fmaf(xr[s], A, Bc);
}

// ---------------- 4. weight f32 -> f16 conversion --------------------------
__global__ __launch_bounds__(256) void wconv_kernel(
    const float* __restrict__ qkv_w, const float* __restrict__ out_w,
    _Float16* __restrict__ wq, _Float16* __restrict__ wo) {
  const int i = blockIdx.x * 256 + threadIdx.x;   // 0..262143
  if (i < 3 * DA * Cc) wq[i] = (_Float16)qkv_w[i];
  else                 wo[i - 3 * DA * Cc] = (_Float16)out_w[i - 3 * DA * Cc];
}

// ---------------- 5. QKV GEMM: (B,S,C)f16 @ w^T -> Q/K/V [b][h][s][d] ------
__global__ __launch_bounds__(256) void qkv_gemm_kernel(
    const _Float16* __restrict__ xn, const _Float16* __restrict__ wq,
    const float* __restrict__ qkvb, _Float16* __restrict__ Qb,
    _Float16* __restrict__ Kb, _Float16* __restrict__ Vb) {
  const int tid = threadIdx.x, lane = tid & 31, wave = tid >> 5;
  const int wl = blockIdx.x * 8 + wave;   // 0..49151
  const int nt = wl % 48;
  const int tmp = wl / 48;
  const int mt = tmp & 63;
  const int b  = tmp >> 6;
  const _Float16* Abase = xn + ((size_t)b * S + mt * 16) * Cc;     // ld=256
  const _Float16* Bbase = wq + (size_t)nt * 16 * Cc;               // ld=256
  v8f acc = {};
#pragma unroll
  for (int kk = 0; kk < Cc; kk += 32) {
    v16h a = load_a_frag(Abase + kk, Cc, lane);
    v16h w = load_b_cont(Bbase + kk, Cc, lane);
    acc = WMMA_F16(a, w, acc);
  }
  const int hf = lane >> 4;
  const int n = nt * 16 + (lane & 15);       // 0..767
  const float bv = qkvb[n];
  const int which = n >> 8;
  const int hh = (n >> 5) & 7;
  const int d  = n & 31;
  _Float16* dst = (which == 0) ? Qb : (which == 1) ? Kb : Vb;
  const size_t base = (size_t)(b * NH + hh) * S * DK + d;
#pragma unroll
  for (int r = 0; r < 8; ++r) {
    const int s = mt * 16 + r + 8 * hf;
    dst[base + (size_t)s * DK] = (_Float16)(acc[r] + bv);
  }
}

// ---------------- 6. flash attention per (b,h) -----------------------------
__global__ __launch_bounds__(256) void attn_kernel(
    const _Float16* __restrict__ Qb, const _Float16* __restrict__ Kb,
    const _Float16* __restrict__ Vb, _Float16* __restrict__ AO) {
  __shared__ __align__(16) _Float16 Kl[128 * DK];
  __shared__ __align__(16) _Float16 Vl[128 * DK];
  __shared__ __align__(16) _Float16 Ps[8][16 * 32];

  const int tid = threadIdx.x, lane = tid & 31, wave = tid >> 5;
  const int n = lane & 15, hf = lane >> 4;
  const int qg = blockIdx.x & 7;
  const int bh = blockIdx.x >> 3;            // 0..127
  const int b = bh >> 3, h = bh & 7;
  const size_t headbase = (size_t)bh * S * DK;

  const int q0 = qg * 128 + wave * 16;
  const v16h qa = load_a_frag(Qb + headbase + (size_t)q0 * DK, DK, lane);

  v8f o0 = {}, o1 = {};
  float m[8], l[8];
#pragma unroll
  for (int r = 0; r < 8; ++r) { m[r] = -3.0e38f; l[r] = 0.f; }
  const float scale = 0.17677669529663687f;  // 1/sqrt(32)

  for (int kc = 0; kc < S; kc += 128) {
    __syncthreads();
#if USE_TDM
    if (tid < 32) {  // wave 0 issues the TDM DMAs (EXEC ignored, per-wave op)
      tdm_load_1d_f16((unsigned)(unsigned long long)(uintptr_t)&Kl[0],
                      (unsigned long long)(uintptr_t)(Kb + headbase + (size_t)kc * DK),
                      128 * DK);
      tdm_load_1d_f16((unsigned)(unsigned long long)(uintptr_t)&Vl[0],
                      (unsigned long long)(uintptr_t)(Vb + headbase + (size_t)kc * DK),
                      128 * DK);
      __builtin_amdgcn_s_wait_tensorcnt(0);
    }
#else
    {
      const _Float16* gk = Kb + headbase + (size_t)kc * DK + tid * 16;
      const _Float16* gv = Vb + headbase + (size_t)kc * DK + tid * 16;
      *(v8h*)&Kl[tid * 16]     = *(const v8h*)(gk);
      *(v8h*)&Kl[tid * 16 + 8] = *(const v8h*)(gk + 8);
      *(v8h*)&Vl[tid * 16]     = *(const v8h*)(gv);
      *(v8h*)&Vl[tid * 16 + 8] = *(const v8h*)(gv + 8);
    }
#endif
    if (kc + 128 < S) {   // global_prefetch_b8 next chunk
      __builtin_prefetch(Kb + headbase + (size_t)(kc + 128) * DK + tid * 16, 0, 1);
      __builtin_prefetch(Vb + headbase + (size_t)(kc + 128) * DK + tid * 16, 0, 1);
    }
    __syncthreads();
#pragma unroll
    for (int j = 0; j < 128; j += 32) {
      // scores: Q(16x32) @ K^T(32x16) twice -> 16x32 score tile
      v16h kb0 = load_b_cont(&Kl[(size_t)j * DK], DK, lane);
      v16h kb1 = load_b_cont(&Kl[(size_t)(j + 16) * DK], DK, lane);
      v8f z = {};
      v8f s0 = WMMA_F16(qa, kb0, z);
      v8f s1 = WMMA_F16(qa, kb1, z);
#pragma unroll
      for (int r = 0; r < 8; ++r) {
        float a0 = s0[r] * scale, a1 = s1[r] * scale;
        float t = fmaxf(a0, a1);
        t = fmaxf(t, __shfl_xor(t, 8));
        t = fmaxf(t, __shfl_xor(t, 4));
        t = fmaxf(t, __shfl_xor(t, 2));
        t = fmaxf(t, __shfl_xor(t, 1));
        const float mn = fmaxf(m[r], t);
        const float al = __expf(m[r] - mn);
        const float p0 = __expf(a0 - mn);
        const float p1 = __expf(a1 - mn);
        float rs = p0 + p1;
        rs += __shfl_xor(rs, 8);
        rs += __shfl_xor(rs, 4);
        rs += __shfl_xor(rs, 2);
        rs += __shfl_xor(rs, 1);
        l[r] = l[r] * al + rs;
        m[r] = mn;
        o0[r] *= al; o1[r] *= al;
        // C-layout (M=r+8*hf, N=n) -> row-major P scratch in LDS
        Ps[wave][(r + 8 * hf) * 32 + n]      = (_Float16)p0;
        Ps[wave][(r + 8 * hf) * 32 + 16 + n] = (_Float16)p1;
      }
      // re-layout P as an A fragment and accumulate P(16x32) @ V(32x32)
      v16h pa  = load_a_frag(&Ps[wave][0], 32, lane);
      v16h vb0 = load_b_rowmajor(&Vl[(size_t)j * DK], DK, lane);       // dk 0..15
      v16h vb1 = load_b_rowmajor(&Vl[(size_t)j * DK + 16], DK, lane);  // dk 16..31
      o0 = WMMA_F16(pa, vb0, o0);
      o1 = WMMA_F16(pa, vb1, o1);
    }
  }
#pragma unroll
  for (int r = 0; r < 8; ++r) {
    const float inv = 1.0f / l[r];
    const int s = q0 + r + 8 * hf;
    const size_t rowo = ((size_t)b * S + s) * DA + h * DK;
    AO[rowo + n]      = (_Float16)(o0[r] * inv);
    AO[rowo + 16 + n] = (_Float16)(o1[r] * inv);
  }
}

// ---------------- 7. output projection + bias + residual -------------------
__global__ __launch_bounds__(256) void out_proj_kernel(
    const _Float16* __restrict__ ao, const _Float16* __restrict__ wo,
    const float* __restrict__ ob, const float* __restrict__ x,
    float* __restrict__ out) {
  const int tid = threadIdx.x, lane = tid & 31, wave = tid >> 5;
  const int wl = blockIdx.x * 8 + wave;     // 0..16383
  const int nt = wl & 15;
  const int tmp = wl >> 4;
  const int mt = tmp & 63;
  const int b  = tmp >> 6;
  const _Float16* Abase = ao + ((size_t)b * S + mt * 16) * DA;   // ld=256
  const _Float16* Bbase = wo + (size_t)nt * 16 * Cc;             // ld=256
  v8f acc = {};
#pragma unroll
  for (int kk = 0; kk < DA; kk += 32) {
    v16h a = load_a_frag(Abase + kk, DA, lane);
    v16h w = load_b_cont(Bbase + kk, Cc, lane);
    acc = WMMA_F16(a, w, acc);
  }
  const int hf = lane >> 4;
  const int c = nt * 16 + (lane & 15);
  const float bv = ob[c];
  const size_t rowbase = ((size_t)b * Cc + c) * S;
#pragma unroll
  for (int r = 0; r < 8; ++r) {
    const int s = mt * 16 + r + 8 * hf;
    out[rowbase + s] = acc[r] + bv + x[rowbase + s];
  }
}

// ---------------- host launcher --------------------------------------------
extern "C" void kernel_launch(void* const* d_in, const int* in_sizes, int n_in,
                              void* d_out, int out_size, void* d_ws, size_t ws_size,
                              hipStream_t stream) {
  const float* x     = (const float*)d_in[0];
  const float* t_emb = (const float*)d_in[1];
  const float* cond  = (const float*)d_in[2];
  const float* gnw   = (const float*)d_in[3];
  const float* gnb   = (const float*)d_in[4];
  const float* pw    = (const float*)d_in[5];
  const float* pb    = (const float*)d_in[6];
  const float* qw    = (const float*)d_in[7];
  const float* qb    = (const float*)d_in[8];
  const float* ow    = (const float*)d_in[9];
  const float* ob    = (const float*)d_in[10];
  float* out = (float*)d_out;

  char* w = (char*)d_ws;
  auto carve = [&](size_t bytes) -> void* {
    void* p = (void*)w;
    w += (bytes + 255) & ~(size_t)255;
    return p;
  };
  float*    gamma = (float*)carve((size_t)Bz * Cc * 4);
  float*    beta  = (float*)carve((size_t)Bz * Cc * 4);
  float*    gnm   = (float*)carve((size_t)Bz * 8 * 4);
  float*    gnr   = (float*)carve((size_t)Bz * 8 * 4);
  _Float16* xn    = (_Float16*)carve((size_t)Bz * S * Cc * 2);
  _Float16* wq16  = (_Float16*)carve((size_t)3 * DA * Cc * 2);
  _Float16* wo16  = (_Float16*)carve((size_t)Cc * Cc * 2);
  _Float16* Qb    = (_Float16*)carve((size_t)Bz * NH * S * DK * 2);
  _Float16* Kb    = (_Float16*)carve((size_t)Bz * NH * S * DK * 2);
  _Float16* Vb    = (_Float16*)carve((size_t)Bz * NH * S * DK * 2);
  _Float16* AO    = (_Float16*)carve((size_t)Bz * S * DA * 2);

  adaln_kernel<<<Bz, 512, 0, stream>>>(t_emb, cond, pw, pb, gamma, beta);
  gn_stats_kernel<<<Bz * 8, 256, 0, stream>>>(x, gnm, gnr);
  gn_norm_kernel<<<Bz * Cc, 256, 0, stream>>>(x, gnw, gnb, gamma, beta, gnm, gnr, xn);
  wconv_kernel<<<(3 * DA * Cc + Cc * Cc) / 256, 256, 0, stream>>>(qw, ow, wq16, wo16);
  qkv_gemm_kernel<<<(Bz * 64 * 48) / 8, 256, 0, stream>>>(xn, wq16, qb, Qb, Kb, Vb);
  attn_kernel<<<Bz * NH * (S / 128), 256, 0, stream>>>(Qb, Kb, Vb, AO);
  out_proj_kernel<<<(Bz * 64 * 16) / 8, 256, 0, stream>>>(AO, wo16, ob, x, out);
}
